// GRU_D_88252987998726
// MI455X (gfx1250) — matrix operand order repaired and checked by reading, not compiled, not run on hardware
//
#include <hip/hip_runtime.h>

// ---------------------------------------------------------------------------
// GRU-D forward on MI455X (gfx1250, wave32, WMMA).
// - All weights converted (and TRANSPOSED to [N,K]) to bf16 once per launch;
//   every matmul runs through v_wmma_f32_16x16x32_bf16 with fp32 accumulation.
// - Weights (~38MB bf16) are L2-resident (192MB): recurrence is WMMA-bound.
// - Tile staging uses GLOBAL_LOAD_ASYNC_TO_LDS_B128 (ASYNCcnt) so the next
//   k-tile streams into LDS while the current tile's WMMAs execute.
// ---------------------------------------------------------------------------

#define T_  200
#define K_  512
#define B_  1024
#define D_  128
#define H_  2048            // D*ND
#define CATW (2*D_ + H_)    // [X_hat | M | hd] = 2304
#define LOGC 0.9189385332046727f

typedef __bf16 v16bf __attribute__((ext_vector_type(16)));
typedef float  v8f   __attribute__((ext_vector_type(8)));
typedef int    v4i   __attribute__((ext_vector_type(4)));

#if defined(__HIP_DEVICE_COMPILE__) && \
    __has_builtin(__builtin_amdgcn_global_load_async_to_lds_b128) && \
    __has_builtin(__builtin_amdgcn_s_wait_asynccnt)
#define ASYNC_LDS 1
#else
#define ASYNC_LDS 0
#endif

// ======================= WMMA tiled GEMM (bf16 -> f32) ======================
// C[M,N] = act( A[M,K](lda) @ Bt[N,K]^T + bias + Cin ), outputs f32 / bf16.
// Block: 256 threads = 8 wave32s, tile 128x128, k-step 32, double-buffered LDS.
#define BM 128
#define BN 128
#define BKK 32

__global__ __launch_bounds__(256)
void gemm_bf16_kernel(const __bf16* __restrict__ A, int lda,
                      const __bf16* __restrict__ Bt,          // [N,K] row-major
                      const float*  __restrict__ bias,        // [N] or null
                      const float*  __restrict__ Cin,         // [M,N] or null
                      float*        __restrict__ outF,        // [M,N] or null
                      __bf16*       __restrict__ outB,        // [M,N] or null
                      int Mrows, int N, int Kd, int relu)
{
    __shared__ __bf16 lA [2][BM * BKK];   // row-major tile of A
    __shared__ __bf16 lBt[2][BN * BKK];   // rows of B^T (K-contiguous)

    const int tid  = threadIdx.x;
    const int lane = tid & 31;
    const int wid  = tid >> 5;
    const int wm   = wid & 1;     // 2 waves along M (64 rows each)
    const int wn   = wid >> 1;    // 4 waves along N (32 cols each)
    const int m0   = blockIdx.y * BM;
    const int n0   = blockIdx.x * BN;
    const int r16  = lane & 15;

    v8f acc[4][2];
    const v8f zero8 = {0.f,0.f,0.f,0.f,0.f,0.f,0.f,0.f};
#pragma unroll
    for (int i = 0; i < 4; ++i)
#pragma unroll
        for (int j = 0; j < 2; ++j) acc[i][j] = zero8;

    const __bf16* Abase = A  + (size_t)m0 * lda;
    const __bf16* Bbase = Bt + (size_t)n0 * Kd;

    // Stage one 128x32 bf16 tile: 512 chunks of 16B, 2 per thread.
    auto stage = [&](const __bf16* src, int src_ld, __bf16* dst, int kb) {
#pragma unroll
        for (int it = 0; it < 2; ++it) {
            int linear = tid + it * 256;
            int row = linear >> 2;
            int cc  = (linear & 3) * 8;
            const __bf16* s = src + (size_t)row * src_ld + kb + cc;
            __bf16* d = dst + row * BKK + cc;
#if ASYNC_LDS
            __builtin_amdgcn_global_load_async_to_lds_b128((v4i*)s, (v4i*)d, 0, 0);
#else
            *(uint4*)d = *(const uint4*)s;
#endif
        }
    };

    stage(Abase, lda, lA[0],  0);
    stage(Bbase, Kd,  lBt[0], 0);
#if ASYNC_LDS
    __builtin_amdgcn_s_wait_asynccnt(0);
#endif
    __syncthreads();

    int cur = 0;
    const int nk = Kd / BKK;
    for (int kt = 0; kt < nk; ++kt) {
        if (kt + 1 < nk) {                      // async prefetch of next tile
            stage(Abase, lda, lA[cur ^ 1],  (kt + 1) * BKK);
            stage(Bbase, Kd,  lBt[cur ^ 1], (kt + 1) * BKK);
        }

        // A fragment (16x32 bf16): lanes 0-15 hold K 0..7 & 16..23,
        // lanes 16-31 hold K 8..15 & 24..31  -> two 16B LDS reads.
        v16bf af[4], bfr[2];
        const int kbA = (lane >> 4) * 8;
#pragma unroll
        for (int i = 0; i < 4; ++i) {
            const __bf16* p = &lA[cur][(wm * 64 + i * 16 + r16) * BKK];
            *((uint4*)&af[i])     = *(const uint4*)(p + kbA);
            *((uint4*)&af[i] + 1) = *(const uint4*)(p + kbA + 16);
        }
        // B fragment (32x16): lane = column; lanes 0-15 K 0..7 & 16..23, 16-31 K 8..15 & 24..31
        const int kbB = (lane >> 4) * 8;
#pragma unroll
        for (int j = 0; j < 2; ++j) {
            const __bf16* p = &lBt[cur][(wn * 32 + j * 16 + r16) * BKK + kbB];
            *((uint4*)&bfr[j])     = *(const uint4*)(p);
            *((uint4*)&bfr[j] + 1) = *(const uint4*)(p + 16);
        }
#pragma unroll
        for (int i = 0; i < 4; ++i)
#pragma unroll
            for (int j = 0; j < 2; ++j)
                acc[i][j] = __builtin_amdgcn_wmma_f32_16x16x32_bf16(
                    false, af[i], false, bfr[j], (short)0, acc[i][j], false, false);

#if ASYNC_LDS
        if (kt + 1 < nk) __builtin_amdgcn_s_wait_asynccnt(0);
#endif
        __syncthreads();
        cur ^= 1;
    }

    // C/D layout: VGPR r -> M = r (+8 for lanes 16-31), N = lane&15
#pragma unroll
    for (int i = 0; i < 4; ++i)
#pragma unroll
        for (int j = 0; j < 2; ++j) {
            const int rbase = m0 + wm * 64 + i * 16 + ((lane >> 4) << 3);
            const int col   = n0 + wn * 32 + j * 16 + r16;
#pragma unroll
            for (int r = 0; r < 8; ++r) {
                size_t o = (size_t)(rbase + r) * N + col;
                float v = acc[i][j][r];
                if (Cin)  v += Cin[o];
                if (bias) v += bias[col];
                if (relu) v = fmaxf(v, 0.f);
                if (outF) outF[o] = v;
                if (outB) outB[o] = (__bf16)v;
            }
        }
}

// ============================ setup / pack kernels ==========================
__global__ void fill0_kernel(float* p, int n) {
    int i = blockIdx.x * blockDim.x + threadIdx.x;
    if (i < n) p[i] = 0.f;
}
// transpose-convert: src f32 [K,N] -> dst bf16 [N,K]
__global__ void cvt_t_kernel(const float* s, __bf16* d, int Krows, int N) {
    int i = blockIdx.x * blockDim.x + threadIdx.x;
    if (i >= Krows * N) return;
    int k = i / N, n = i % N;
    d[(size_t)n * Krows + k] = (__bf16)s[i];
}
// B^T pack of [W_r|W_z ; V_r|V_z ; U_r|U_z] -> out[col(2H)][row(CATW)]
__global__ void pack_rz_t_kernel(const float* Wr, const float* Wz,
                                 const float* Vr, const float* Vz,
                                 const float* Ur, const float* Uz, __bf16* out) {
    int i = blockIdx.x * blockDim.x + threadIdx.x;
    const int NC = 2 * H_;
    if (i >= CATW * NC) return;
    int row = i / NC, col = i % NC;
    int side = col / H_, c = col % H_;
    float v;
    if (row < D_)          v = (side ? Wz : Wr)[row * H_ + c];
    else if (row < 2 * D_) v = (side ? Vz : Vr)[(row - D_) * H_ + c];
    else                   v = (side ? Uz : Ur)[(row - 2 * D_) * H_ + c];
    out[(size_t)col * CATW + row] = (__bf16)v;
}
// B^T pack of [W_h ; V_h] -> out[col(H)][row(2D)]
__global__ void pack_h1_t_kernel(const float* Wh, const float* Vh, __bf16* out) {
    int i = blockIdx.x * blockDim.x + threadIdx.x;
    if (i >= 2 * D_ * H_) return;
    int row = i / H_, c = i % H_;
    float v = (row < D_) ? Wh[row * H_ + c] : Vh[(row - D_) * H_ + c];
    out[(size_t)c * (2 * D_) + row] = (__bf16)v;
}
__global__ void pack2_kernel(const float* a, const float* b, float* out, int n) {
    int i = blockIdx.x * blockDim.x + threadIdx.x;
    if (i < 2 * n) out[i] = (i < n) ? a[i] : b[i - n];
}

// ============================ per-step kernels ==============================
__global__ __launch_bounds__(256)
void colsum_kernel(const float* Xo, const float* Mo, float* sumX, float* sumM) {
    __shared__ float sx[256], sm[256];
    int d = blockIdx.x, t = threadIdx.x;
    float ax = 0.f, am = 0.f;
    for (int k = t; k < K_; k += 256) {
        ax += Xo[k * D_ + d];
        am += Mo[k * D_ + d] + 1e-6f;
    }
    sx[t] = ax; sm[t] = am; __syncthreads();
    for (int s = 128; s > 0; s >>= 1) {
        if (t < s) { sx[t] += sx[t + s]; sm[t] += sm[t + s]; }
        __syncthreads();
    }
    if (t == 0) { sumX[d] = sx[0]; sumM[d] = sm[0]; }
}

__global__ void pre1_kernel(const float* tptr, const float* Xo, const float* Mo,
                            const int* idx, float* last_x, float* last_t,
                            __bf16* ivb, float* lxbuf) {
    int i = blockIdx.x * blockDim.x + threadIdx.x;
    if (i >= K_ * D_) return;
    int k = i / D_, d = i % D_;
    float t = *tptr;
    float m = Mo[i], x = Xo[i];
    int b = idx[k];
    float plx = last_x[b * D_ + d];
    float lx = plx * (1.f - m) + x * m;
    last_x[b * D_ + d] = lx;
    float ltg = last_t[b * D_ + d];
    float iv = t - ltg;
    last_t[b * D_ + d] = ltg * (1.f - m) + t * m;
    ivb[i] = (__bf16)iv;
    lxbuf[i] = lx;
}

__global__ void pre2_kernel(const float* Xo, const float* Mo, const float* gx_pre,
                            const float* lxbuf, const float* sumX, const float* sumM,
                            __bf16* in_cat) {
    int i = blockIdx.x * blockDim.x + threadIdx.x;
    if (i >= K_ * D_) return;
    int k = i / D_, d = i % D_;
    float gx = expf(-fmaxf(0.f, gx_pre[i]));
    float meanx = sumX[d] / sumM[d];
    float m = Mo[i], x = Xo[i];
    float xhat = m * x + (1.f - m) * gx * lxbuf[i] + (1.f - m) * (1.f - gx) * meanx;
    in_cat[(size_t)k * CATW + d]       = (__bf16)xhat;
    in_cat[(size_t)k * CATW + D_ + d]  = (__bf16)m;
}

__global__ void pre3_kernel(const float* h, const int* idx, const float* gh_pre,
                            __bf16* hgb, float* hd, __bf16* in_cat) {
    int i = blockIdx.x * blockDim.x + threadIdx.x;
    if (i >= K_ * H_) return;
    int k = i / H_, j = i % H_;
    float hgv = h[(size_t)idx[k] * H_ + j];
    float gh = expf(-fmaxf(0.f, gh_pre[i]));
    float hdv = gh * hgv;
    hgb[i] = (__bf16)hgv;
    hd[i] = hdv;
    in_cat[(size_t)k * CATW + 2 * D_ + j] = (__bf16)hdv;
}

__global__ __launch_bounds__(256)
void loss_kernel(const float* Xo, const float* Mo, const float* p, float* acc) {
    __shared__ float sl[256], sm[256];
    int t = threadIdx.x;
    int i = blockIdx.x * 256 + t;
    float l = 0.f, mm = 0.f;
    if (i < K_ * D_) {
        int k = i / D_, d = i % D_;
        float mean = p[k * 2 * D_ + d];
        float logv = p[k * 2 * D_ + D_ + d];
        float m = Mo[i];
        float err = (Xo[i] - mean) * expf(-0.5f * logv);
        l = 0.5f * ((err * err + logv + 2.f * LOGC) * m);
        mm = m;
    }
    sl[t] = l; sm[t] = mm; __syncthreads();
    for (int s = 128; s > 0; s >>= 1) {
        if (t < s) { sl[t] += sl[t + s]; sm[t] += sm[t + s]; }
        __syncthreads();
    }
    if (t == 0) { atomicAdd(&acc[0], sl[0]); atomicAdd(&acc[1], sm[0]); }
}

__global__ void rz_kernel(const float* Grz, const float* hd, __bf16* rhdb, float* zbuf) {
    int i = blockIdx.x * blockDim.x + threadIdx.x;
    if (i >= K_ * H_) return;
    int k = i / H_, j = i % H_;
    float r = 1.f / (1.f + expf(-Grz[(size_t)k * 2 * H_ + j]));
    float z = 1.f / (1.f + expf(-Grz[(size_t)k * 2 * H_ + H_ + j]));
    rhdb[i] = (__bf16)(r * hd[i]);
    zbuf[i] = z;
}

__global__ void final_kernel(const float* Gh, const float* zbuf, const float* hd,
                             const int* idx, float* h) {
    int i = blockIdx.x * blockDim.x + threadIdx.x;
    if (i >= K_ * H_) return;
    int k = i / H_, j = i % H_;
    float ht = tanhf(Gh[i]);
    float z = zbuf[i];
    h[(size_t)idx[k] * H_ + j] = (1.f - z) * hd[i] + z * ht;
}

__global__ void finalize_kernel(const float* acc, float* out) {
    if (blockIdx.x == 0 && threadIdx.x == 0) out[0] = acc[0] / acc[1];
}

// ================================ host side =================================
extern "C" void kernel_launch(void* const* d_in, const int* in_sizes, int n_in,
                              void* d_out, int out_size, void* d_ws, size_t ws_size,
                              hipStream_t stream) {
    const float* obs_times = (const float*)d_in[0];
    const float* X   = (const float*)d_in[1];
    const float* M   = (const float*)d_in[2];
    const float* Wp1 = (const float*)d_in[3];
    const float* bp1 = (const float*)d_in[4];
    const float* Wp2 = (const float*)d_in[5];
    const float* bp2 = (const float*)d_in[6];
    const float* W_r = (const float*)d_in[7];
    const float* V_r = (const float*)d_in[8];
    const float* U_r = (const float*)d_in[9];
    const float* W_z = (const float*)d_in[10];
    const float* V_z = (const float*)d_in[11];
    const float* U_z = (const float*)d_in[12];
    const float* W_h = (const float*)d_in[13];
    const float* V_h = (const float*)d_in[14];
    const float* U_h = (const float*)d_in[15];
    const float* b_r = (const float*)d_in[16];
    const float* b_z = (const float*)d_in[17];
    const float* b_h = (const float*)d_in[18];
    const float* Wgx = (const float*)d_in[19];
    const float* Wgh = (const float*)d_in[20];
    const int*   batch_idx = (const int*)d_in[21];

    char* base = (char*)d_ws;
    size_t off = 0;
    auto alloc = [&](size_t bytes) -> void* {
        off = (off + 255) & ~(size_t)255;
        void* p = base + off;
        off += bytes;
        return p;
    };

    // bf16 transposed weights ([N,K] row-major)
    __bf16* Wp1t = (__bf16*)alloc((size_t)H_ * H_ * 2);
    __bf16* Wp2t = (__bf16*)alloc((size_t)H_ * 2 * D_ * 2);
    __bf16* Wgxt = (__bf16*)alloc((size_t)D_ * D_ * 2);
    __bf16* Wght = (__bf16*)alloc((size_t)D_ * H_ * 2);
    __bf16* Uht  = (__bf16*)alloc((size_t)H_ * H_ * 2);
    __bf16* Wrzt = (__bf16*)alloc((size_t)CATW * 2 * H_ * 2);
    __bf16* Wh1t = (__bf16*)alloc((size_t)2 * D_ * H_ * 2);
    float*  brz  = (float*)alloc((size_t)2 * H_ * 4);
    // persistent state
    float* h      = (float*)alloc((size_t)B_ * H_ * 4);
    float* last_x = (float*)alloc((size_t)B_ * D_ * 4);
    float* last_t = (float*)alloc((size_t)B_ * D_ * 4);
    float* acc    = (float*)alloc(2 * 4);
    // per-step temporaries
    float*  sumX   = (float*)alloc(D_ * 4);
    float*  sumM   = (float*)alloc(D_ * 4);
    __bf16* ivb    = (__bf16*)alloc((size_t)K_ * D_ * 2);
    float*  lxbuf  = (float*)alloc((size_t)K_ * D_ * 4);
    float*  gx_pre = (float*)alloc((size_t)K_ * D_ * 4);
    float*  gh_pre = (float*)alloc((size_t)K_ * H_ * 4);
    __bf16* in_cat = (__bf16*)alloc((size_t)K_ * CATW * 2);
    __bf16* hgb    = (__bf16*)alloc((size_t)K_ * H_ * 2);
    float*  hd     = (float*)alloc((size_t)K_ * H_ * 4);
    __bf16* p1b    = (__bf16*)alloc((size_t)K_ * H_ * 2);
    float*  pbuf   = (float*)alloc((size_t)K_ * 2 * D_ * 4);
    float*  Grz    = (float*)alloc((size_t)K_ * 2 * H_ * 4);
    float*  Gh     = (float*)alloc((size_t)K_ * H_ * 4);
    float*  zbuf   = (float*)alloc((size_t)K_ * H_ * 4);
    __bf16* rhdb   = (__bf16*)alloc((size_t)K_ * H_ * 2);
    (void)ws_size; (void)in_sizes; (void)n_in; (void)out_size;

    auto cdiv = [](int a, int b) { return (a + b - 1) / b; };
    auto gemm = [&](const __bf16* A, int lda, const __bf16* Bt, const float* bias,
                    const float* Cin, float* oF, __bf16* oB, int N, int Kd, int relu) {
        dim3 g(N / BN, K_ / BM);
        gemm_bf16_kernel<<<g, dim3(256), 0, stream>>>(A, lda, Bt, bias, Cin, oF, oB,
                                                      K_, N, Kd, relu);
    };

    // --------------------- one-time setup (per launch) ----------------------
    cvt_t_kernel<<<cdiv(H_*H_,256),256,0,stream>>>(Wp1, Wp1t, H_, H_);
    cvt_t_kernel<<<cdiv(H_*2*D_,256),256,0,stream>>>(Wp2, Wp2t, H_, 2*D_);
    cvt_t_kernel<<<cdiv(D_*D_,256),256,0,stream>>>(Wgx, Wgxt, D_, D_);
    cvt_t_kernel<<<cdiv(D_*H_,256),256,0,stream>>>(Wgh, Wght, D_, H_);
    cvt_t_kernel<<<cdiv(H_*H_,256),256,0,stream>>>(U_h, Uht, H_, H_);
    pack_rz_t_kernel<<<cdiv(CATW*2*H_,256),256,0,stream>>>(W_r,W_z,V_r,V_z,U_r,U_z,Wrzt);
    pack_h1_t_kernel<<<cdiv(2*D_*H_,256),256,0,stream>>>(W_h, V_h, Wh1t);
    pack2_kernel<<<cdiv(2*H_,256),256,0,stream>>>(b_r, b_z, brz, H_);
    fill0_kernel<<<cdiv(B_*H_,256),256,0,stream>>>(h, B_*H_);
    fill0_kernel<<<cdiv(B_*D_,256),256,0,stream>>>(last_x, B_*D_);
    fill0_kernel<<<cdiv(B_*D_,256),256,0,stream>>>(last_t, B_*D_);
    fill0_kernel<<<1,256,0,stream>>>(acc, 2);

    const int KD = K_ * D_, KH = K_ * H_;

    // --------------------------- recurrence --------------------------------
    for (int step = 0; step < T_; ++step) {
        const float* Xo   = X + (size_t)step * KD;
        const float* Mo   = M + (size_t)step * KD;
        const int*   idx  = batch_idx + (size_t)step * K_;
        const float* tptr = obs_times + step;

        colsum_kernel<<<D_,256,0,stream>>>(Xo, Mo, sumX, sumM);
        pre1_kernel<<<cdiv(KD,256),256,0,stream>>>(tptr, Xo, Mo, idx, last_x, last_t,
                                                   ivb, lxbuf);
        // decay pre-activations: iv @ Wgx, iv @ Wgh
        gemm(ivb, D_, Wgxt, nullptr, nullptr, gx_pre, nullptr, D_, D_, 0);
        gemm(ivb, D_, Wght, nullptr, nullptr, gh_pre, nullptr, H_, D_, 0);
        pre2_kernel<<<cdiv(KD,256),256,0,stream>>>(Xo, Mo, gx_pre, lxbuf, sumX, sumM,
                                                   in_cat);
        pre3_kernel<<<cdiv(KH,256),256,0,stream>>>(h, idx, gh_pre, hgb, hd, in_cat);
        // p-model: relu(hg@Wp1+bp1)@Wp2+bp2
        gemm(hgb, H_, Wp1t, bp1, nullptr, nullptr, p1b, H_, H_, 1);
        gemm(p1b, H_, Wp2t, bp2, nullptr, pbuf, nullptr, 2*D_, H_, 0);
        loss_kernel<<<cdiv(KD,256),256,0,stream>>>(Xo, Mo, pbuf, acc);
        // fused r/z gates: [X_hat|M|hd] @ [W|V|U] + [b_r|b_z]
        gemm(in_cat, CATW, Wrzt, brz, nullptr, Grz, nullptr, 2*H_, CATW, 0);
        // h-tilde partial: [X_hat|M] @ [W_h;V_h] + b_h
        gemm(in_cat, CATW, Wh1t, b_h, nullptr, Gh, nullptr, H_, 2*D_, 0);
        rz_kernel<<<cdiv(KH,256),256,0,stream>>>(Grz, hd, rhdb, zbuf);
        // h-tilde: += (r*hd) @ U_h  (Cin accumulate, in place)
        gemm(rhdb, H_, Uht, nullptr, Gh, Gh, nullptr, H_, H_, 0);
        final_kernel<<<cdiv(KH,256),256,0,stream>>>(Gh, zbuf, hd, idx, h);
    }

    finalize_kernel<<<1,1,0,stream>>>(acc, (float*)d_out);
}
